// RoPEMultiheadSelfAttention_55748675502795
// MI455X (gfx1250) — matrix operand order repaired and checked by reading, not compiled, not run on hardware
//
#include <hip/hip_runtime.h>
#include <hip/hip_bf16.h>

// ---------------------------------------------------------------------------
// RoPE multi-head self-attention, bf16 WMMA pipeline for gfx1250 (MI455X).
// B=4, S=2048, E=1024, H=16, D=64.  fp32 in/out, bf16 WMMA w/ fp32 accum.
// ---------------------------------------------------------------------------

typedef __attribute__((ext_vector_type(16))) __bf16 v16bf;
typedef __attribute__((ext_vector_type(8)))  __bf16 v8bf;
typedef __attribute__((ext_vector_type(8)))  float  v8f;

constexpr int Bb = 4;
constexpr int Ss = 2048;
constexpr int Ee = 1024;
constexpr int Hh = 16;
constexpr int Dd = 64;
constexpr int Mrows = Bb * Ss;          // 8192
constexpr int PLDS_STRIDE = 72;         // padded row stride (bf16 elems), 144B = 9*16B

// ---- fragment loaders -----------------------------------------------------
// A-matrix 16x32 bf16 per-lane layout: lanes 0-15 -> K {k0..k0+7, k0+16..k0+23},
// lanes 16-31 -> K {k0+8..k0+15, k0+24..k0+31}. Two 16B loads.
__device__ __forceinline__ v16bf ld_fragA(const __bf16* row, int k0, int half) {
  const __bf16* p = row + k0 + half * 8;
  union { v16bf v; v8bf h[2]; } u;
  u.h[0] = *(const v8bf*)(p);
  u.h[1] = *(const v8bf*)(p + 16);
  return u.v;
}
// B-matrix 32x16 bf16 per-lane layout: lane n holds column n, lanes 0-15 K {k0..k0+7, k0+16..k0+23},
// lanes 16-31 K {k0+8..k0+15, k0+24..k0+31}. Two 16B loads.
__device__ __forceinline__ v16bf ld_fragB(const __bf16* row, int k0, int half) {
  const __bf16* p = row + k0 + half * 8;
  union { v16bf v; v8bf h[2]; } u;
  u.h[0] = *(const v8bf*)(p);
  u.h[1] = *(const v8bf*)(p + 16);
  return u.v;
}

// ---- fp32 -> bf16 convert -------------------------------------------------
__global__ void cvt_f32_bf16(const float* __restrict__ in, __bf16* __restrict__ out, int n) {
  int i = blockIdx.x * blockDim.x + threadIdx.x;
  if (i < n) out[i] = (__bf16)in[i];
}

// ---- GEMM: C[m,n] = sum_k A[m,k] * W[n,k] + bias[n]  (x @ W.T + b) --------
// A: bf16 [M][K] row-major, W: bf16 [N][K] row-major, C: fp32 [M][N].
// Block = 8 waves; block tile 128(M) x 128(N); wave tile 32x64 = 2x4 WMMA frags
// (8 WMMA per 12 b128 loads per K-step; A reused x4, B reused x2 in registers).
__global__ __launch_bounds__(256) void gemm_xwT_bias(
    const __bf16* __restrict__ A, const __bf16* __restrict__ W,
    const float* __restrict__ bias, float* __restrict__ C,
    int M, int N, int K) {
  const int w    = threadIdx.x >> 5;
  const int lane = threadIdx.x & 31;
  const int half = lane >> 4;
  const int lm   = lane & 15;
  const int mtile = blockIdx.y * 128;
  const int ntile = blockIdx.x * 128;
  const int wm = w & 3;                 // 4 waves along M, 32 rows each
  const int wn = w >> 2;                // 2 waves along N, 64 cols each
  const int rowA0 = mtile + wm * 32;
  const int colB0 = ntile + wn * 64;

  v8f acc[2][4];
  v8f z = {};
  #pragma unroll
  for (int mi = 0; mi < 2; ++mi)
    #pragma unroll
    for (int ni = 0; ni < 4; ++ni) acc[mi][ni] = z;

  const __bf16* arow[2];
  arow[0] = A + (size_t)(rowA0 + lm) * K;
  arow[1] = A + (size_t)(rowA0 + 16 + lm) * K;
  const __bf16* brow[4];
  #pragma unroll
  for (int ni = 0; ni < 4; ++ni)
    brow[ni] = W + (size_t)(colB0 + ni * 16 + lm) * K;

  for (int kk = 0; kk < K; kk += 32) {
    v16bf af[2], bfr[4];
    #pragma unroll
    for (int mi = 0; mi < 2; ++mi) af[mi] = ld_fragA(arow[mi], kk, half);
    #pragma unroll
    for (int ni = 0; ni < 4; ++ni) bfr[ni] = ld_fragB(brow[ni], kk, half);
    #pragma unroll
    for (int mi = 0; mi < 2; ++mi)
      #pragma unroll
      for (int ni = 0; ni < 4; ++ni)
        acc[mi][ni] = __builtin_amdgcn_wmma_f32_16x16x32_bf16(
            false, af[mi], false, bfr[ni], (short)0, acc[mi][ni], false, false);
  }

  #pragma unroll
  for (int mi = 0; mi < 2; ++mi) {
    #pragma unroll
    for (int ni = 0; ni < 4; ++ni) {
      const int col = colB0 + ni * 16 + lm;
      const float bv = bias[col];
      const int row0 = rowA0 + mi * 16 + half * 8;
      #pragma unroll
      for (int r = 0; r < 8; ++r) {
        C[(size_t)(row0 + r) * N + col] = acc[mi][ni][r] + bv;
      }
    }
  }
}

// ---- RoPE + repack to head-major bf16 (B,H,S,D) ---------------------------
// in: fp32 (B,S,H*D); out: bf16 [(b*H+h)*S + s][d]; scale pre-applied (q: D^-0.5).
__global__ void rope_pack(const float* __restrict__ in, __bf16* __restrict__ out, float scale) {
  int idx = blockIdx.x * blockDim.x + threadIdx.x;   // B*H*S*32 threads
  int d  = idx & 31;
  int s  = (idx >> 5) & (Ss - 1);
  int bh = idx >> 16;
  if (bh >= Bb * Hh) return;
  int b = bh >> 4, h = bh & 15;

  const float* pin = in + (size_t)(b * Ss + s) * Ee + h * Dd;
  float x1 = pin[d];
  float x2 = pin[d + 32];

  int i1 = d >> 1;
  int i2 = 16 + (d >> 1);
  float th1 = (float)s * __powf(10000.f, -(float)(2 * i1) * (1.0f / 64.0f));
  float th2 = (float)s * __powf(10000.f, -(float)(2 * i2) * (1.0f / 64.0f));
  float c1 = __cosf(th1), s1 = __sinf(th1);
  float c2 = __cosf(th2), s2 = __sinf(th2);

  __bf16* po = out + (size_t)(bh * Ss + s) * Dd;
  po[d]      = (__bf16)((x1 * c1 - x2 * s1) * scale);   // rotate_half: -x[d+32]
  po[d + 32] = (__bf16)((x2 * c2 + x1 * s2) * scale);   // rotate_half: +x[d]
}

// ---- V repack: fp32 (B,S,H*D) -> bf16 transposed (B,H,D,S) ----------------
__global__ void v_pack_T(const float* __restrict__ in, __bf16* __restrict__ out) {
  int idx = blockIdx.x * blockDim.x + threadIdx.x;   // B*H*D*S threads
  int s  = idx & (Ss - 1);
  int t  = idx >> 11;
  int d  = t & 63;
  int bh = t >> 6;
  if (bh >= Bb * Hh) return;
  int b = bh >> 4, h = bh & 15;
  out[(size_t)(bh * Dd + d) * Ss + s] =
      (__bf16)in[(size_t)(b * Ss + s) * Ee + h * Dd + d];
}

// ---- Flash attention: per (b,h), 128 query rows per block, 16 per wave ----
// Q: bf16 (BH,S,D) pre-scaled; K: bf16 (BH,S,D); Vt: bf16 (BH,D,S).
// Out: bf16 (B,S,E) ready for the output projection GEMM.
__global__ __launch_bounds__(256) void flash_attn(
    const __bf16* __restrict__ Q, const __bf16* __restrict__ K,
    const __bf16* __restrict__ Vt, __bf16* __restrict__ Out) {
  __shared__ __bf16 plds[8 * 16 * PLDS_STRIDE];

  const int bh    = blockIdx.x;
  const int qtile = blockIdx.y * 128;
  const int w     = threadIdx.x >> 5;
  const int lane  = threadIdx.x & 31;
  const int half  = lane >> 4;
  const int lm    = lane & 15;
  const int b = bh >> 4, h = bh & 15;

  const __bf16* Qb  = Q  + (size_t)bh * Ss * Dd;
  const __bf16* Kb  = K  + (size_t)bh * Ss * Dd;
  const __bf16* Vtb = Vt + (size_t)bh * Dd * Ss;

  // Q fragments for this wave's 16 rows (held in registers for the whole loop)
  const __bf16* qrow = Qb + (size_t)(qtile + w * 16 + lm) * Dd;
  v16bf qa0 = ld_fragA(qrow, 0, half);
  v16bf qa1 = ld_fragA(qrow, 32, half);

  float m[8], l[8];
  v8f o[4];
  v8f z = {};
  #pragma unroll
  for (int r = 0; r < 8; ++r) { m[r] = -1e30f; l[r] = 0.f; }
  #pragma unroll
  for (int ni = 0; ni < 4; ++ni) o[ni] = z;

  const int ldbase = w * 16 * PLDS_STRIDE;

  for (int kb = 0; kb < Ss; kb += 64) {
    // ---- scores: S = Q (16x64) x K^T  (4 fragments of 16x16) ----
    v8f sc[4];
    #pragma unroll
    for (int n = 0; n < 4; ++n) sc[n] = z;
    #pragma unroll
    for (int n = 0; n < 4; ++n) {
      const __bf16* krow = Kb + (size_t)(kb + n * 16 + lm) * Dd;
      v16bf kf0 = ld_fragB(krow, 0, half);
      v16bf kf1 = ld_fragB(krow, 32, half);
      sc[n] = __builtin_amdgcn_wmma_f32_16x16x32_bf16(false, qa0, false, kf0, (short)0, sc[n], false, false);
      sc[n] = __builtin_amdgcn_wmma_f32_16x16x32_bf16(false, qa1, false, kf1, (short)0, sc[n], false, false);
    }

    // ---- online softmax: row max over 64 keys ----
    float bm[8];
    #pragma unroll
    for (int r = 0; r < 8; ++r)
      bm[r] = fmaxf(fmaxf(sc[0][r], sc[1][r]), fmaxf(sc[2][r], sc[3][r]));
    #pragma unroll
    for (int off = 1; off < 16; off <<= 1)
      #pragma unroll
      for (int r = 0; r < 8; ++r)
        bm[r] = fmaxf(bm[r], __shfl_xor(bm[r], off, 32));

    float corr[8];
    #pragma unroll
    for (int r = 0; r < 8; ++r) {
      float mn = fmaxf(m[r], bm[r]);
      corr[r] = __expf(m[r] - mn);
      m[r] = mn;
    }

    // P = exp(S - m), per-row partial sums
    float psum[8];
    #pragma unroll
    for (int r = 0; r < 8; ++r) psum[r] = 0.f;
    #pragma unroll
    for (int n = 0; n < 4; ++n)
      #pragma unroll
      for (int r = 0; r < 8; ++r) {
        float p = __expf(sc[n][r] - m[r]);
        sc[n][r] = p;
        psum[r] += p;
      }
    #pragma unroll
    for (int off = 1; off < 16; off <<= 1)
      #pragma unroll
      for (int r = 0; r < 8; ++r)
        psum[r] += __shfl_xor(psum[r], off, 32);
    #pragma unroll
    for (int r = 0; r < 8; ++r) l[r] = l[r] * corr[r] + psum[r];
    #pragma unroll
    for (int ni = 0; ni < 4; ++ni)
      #pragma unroll
      for (int r = 0; r < 8; ++r) o[ni][r] *= corr[r];

    // ---- stage P (C-layout) into LDS, reload in A-layout ----
    asm volatile("" ::: "memory");   // keep prior LDS loads before these stores
    #pragma unroll
    for (int n = 0; n < 4; ++n)
      #pragma unroll
      for (int r = 0; r < 8; ++r)
        plds[ldbase + (r + half * 8) * PLDS_STRIDE + n * 16 + lm] = (__bf16)sc[n][r];
    asm volatile("s_wait_dscnt 0" ::: "memory");  // per-wave DS ordering barrier

    const __bf16* prow = &plds[ldbase + lm * PLDS_STRIDE];
    #pragma unroll
    for (int t = 0; t < 2; ++t) {
      v16bf pa = ld_fragA(prow, t * 32, half);
      #pragma unroll
      for (int ni = 0; ni < 4; ++ni) {
        const __bf16* vrow = Vtb + (size_t)(ni * 16 + lm) * Ss + kb;
        v16bf vf = ld_fragB(vrow, t * 32, half);
        o[ni] = __builtin_amdgcn_wmma_f32_16x16x32_bf16(false, pa, false, vf, (short)0, o[ni], false, false);
      }
    }
  }

  // ---- epilogue: O / l, write to (B,S,E) bf16 ----
  #pragma unroll
  for (int r = 0; r < 8; ++r) {
    const int srow = qtile + w * 16 + r + half * 8;
    const float inv = 1.0f / l[r];
    __bf16* po = Out + (size_t)(b * Ss + srow) * Ee + h * Dd;
    #pragma unroll
    for (int ni = 0; ni < 4; ++ni)
      po[ni * 16 + lm] = (__bf16)(o[ni][r] * inv);
  }
}

// ---------------------------------------------------------------------------
extern "C" void kernel_launch(void* const* d_in, const int* in_sizes, int n_in,
                              void* d_out, int out_size, void* d_ws, size_t ws_size,
                              hipStream_t stream) {
  (void)in_sizes; (void)n_in; (void)out_size; (void)ws_size;
  const float* x  = (const float*)d_in[0];
  const float* wq = (const float*)d_in[1];
  const float* bq = (const float*)d_in[2];
  const float* wk = (const float*)d_in[3];
  const float* bk = (const float*)d_in[4];
  const float* wv = (const float*)d_in[5];
  const float* bv = (const float*)d_in[6];
  const float* wo = (const float*)d_in[7];
  const float* bo = (const float*)d_in[8];

  char* ws = (char*)d_ws;
  size_t off = 0;
  auto take = [&](size_t bytes) { char* p = ws + off; off += bytes; return p; };

  const size_t XB   = (size_t)Mrows * Ee * sizeof(__bf16);      // 16 MB
  const size_t WB   = (size_t)Ee * Ee * sizeof(__bf16);         //  2 MB
  const size_t TMPF = (size_t)Mrows * Ee * sizeof(float);       // 32 MB
  const size_t HB   = (size_t)Mrows * Ee * sizeof(__bf16);      // 16 MB

  __bf16* xb   = (__bf16*)take(XB);
  __bf16* wqb  = (__bf16*)take(WB);
  __bf16* wkb  = (__bf16*)take(WB);
  __bf16* wvb  = (__bf16*)take(WB);
  __bf16* wob  = (__bf16*)take(WB);
  float*  tmpf = (float*) take(TMPF);
  __bf16* qb   = (__bf16*)take(HB);
  __bf16* kbuf = (__bf16*)take(HB);
  __bf16* vtb  = (__bf16*)take(HB);
  __bf16* attb = (__bf16*)take(HB);

  const int nx = Mrows * Ee;      // 8,388,608
  const int nw = Ee * Ee;         // 1,048,576
  cvt_f32_bf16<<<(nx + 255) / 256, 256, 0, stream>>>(x,  xb,  nx);
  cvt_f32_bf16<<<(nw + 255) / 256, 256, 0, stream>>>(wq, wqb, nw);
  cvt_f32_bf16<<<(nw + 255) / 256, 256, 0, stream>>>(wk, wkb, nw);
  cvt_f32_bf16<<<(nw + 255) / 256, 256, 0, stream>>>(wv, wvb, nw);
  cvt_f32_bf16<<<(nw + 255) / 256, 256, 0, stream>>>(wo, wob, nw);

  dim3 ggrid(Ee / 128, Mrows / 128);   // (8, 64)
  const int nrope = Bb * Hh * Ss * 32;    // 4,194,304
  const int nvp   = Bb * Hh * Dd * Ss;    // 8,388,608
  const float qscale = 0.125f;            // D^-0.5, folded into Q

  // Q
  gemm_xwT_bias<<<ggrid, 256, 0, stream>>>(xb, wqb, bq, tmpf, Mrows, Ee, Ee);
  rope_pack<<<(nrope + 255) / 256, 256, 0, stream>>>(tmpf, qb, qscale);
  // K
  gemm_xwT_bias<<<ggrid, 256, 0, stream>>>(xb, wkb, bk, tmpf, Mrows, Ee, Ee);
  rope_pack<<<(nrope + 255) / 256, 256, 0, stream>>>(tmpf, kbuf, 1.0f);
  // V
  gemm_xwT_bias<<<ggrid, 256, 0, stream>>>(xb, wvb, bv, tmpf, Mrows, Ee, Ee);
  v_pack_T<<<(nvp + 255) / 256, 256, 0, stream>>>(tmpf, vtb);

  // Attention
  dim3 agrid(Bb * Hh, Ss / 128);      // (64, 16)
  flash_attn<<<agrid, 256, 0, stream>>>(qb, kbuf, vtb, attb);

  // Output projection -> fp32 d_out
  gemm_xwT_bias<<<ggrid, 256, 0, stream>>>(attb, wob, bo, (float*)d_out, Mrows, Ee, Ee);
}
